// MultiHeadAttention_48850958025227
// MI455X (gfx1250) — compile-verified, matrix operand order checked by reading, not run on hardware
//
#include <hip/hip_runtime.h>

#define S_LEN   4096
#define D_MODEL 1024
#define H_HEADS 16
#define DKV     64

typedef __attribute__((ext_vector_type(16))) __bf16 v16bf;
typedef __attribute__((ext_vector_type(8)))  float  v8f;

union Frag16 {
    v16bf bf;
    unsigned int u[8];
};

__device__ __forceinline__ unsigned short f32_to_bf16(float f) {
    unsigned int u = __float_as_uint(f);
    u += 0x7FFFu + ((u >> 16) & 1u);     // round-to-nearest-even
    return (unsigned short)(u >> 16);
}

// Load a 16x32 bf16 A-fragment (or symmetric B-fragment) from a K-major
// matrix: element (row, k) at p[row*ld + k].  CDNA5 16-bit layout:
// lanes 0-15 rows 0-15; VGPR v holds K-pair kb = (v<4 ? 2v : 16+2(v-4)) + half*8.
__device__ __forceinline__ v16bf load_frag_rowK(const unsigned short* __restrict__ p,
                                                int ld, int row0, int k0) {
    const int lane = threadIdx.x & 31;
    const int r    = lane & 15;
    const int half = lane >> 4;
    Frag16 f;
    const unsigned short* base = p + (size_t)(row0 + r) * ld + k0;
#pragma unroll
    for (int v = 0; v < 8; ++v) {
        int kb = ((v < 4) ? (2 * v) : (16 + 2 * (v - 4))) + half * 8;
        f.u[v] = *(const unsigned int*)(base + kb);
    }
    return f.bf;
}

// CDNA5 async global->LDS copy (16B), tracked by ASYNCcnt.
// vdst = LDS byte offset (low 32 bits of the flat shared address),
// vaddr = 64-bit global address.
__device__ __forceinline__ void async_load_b128(void* lds_ptr, const void* gptr) {
    unsigned lds_off = (unsigned)(unsigned long long)lds_ptr;
    unsigned long long ga = (unsigned long long)gptr;
    asm volatile("global_load_async_to_lds_b128 %0, %1, off"
                 :: "v"(lds_off), "v"(ga) : "memory");
}

// ---------------------------------------------------------------- pre-passes
__global__ void cvt_f32_to_bf16_kernel(const float* __restrict__ x,
                                       unsigned short* __restrict__ y, int n) {
    int i = blockIdx.x * blockDim.x + threadIdx.x;
    int stride = gridDim.x * blockDim.x;
    for (; i < n; i += stride) y[i] = f32_to_bf16(x[i]);
}

// WQ[h, d, k]  ->  Wt[(h*64+k), d]   (K-major for B-fragments)
__global__ void pack_headW_kernel(const float* __restrict__ W,
                                  unsigned short* __restrict__ Wt, int n) {
    int i = blockIdx.x * blockDim.x + threadIdx.x;
    int stride = gridDim.x * blockDim.x;
    for (; i < n; i += stride) {
        int col = i >> 10;              // h*64+k
        int d   = i & 1023;
        int h   = col >> 6;
        int k   = col & 63;
        Wt[i] = f32_to_bf16(W[((size_t)h * D_MODEL + d) * DKV + k]);
    }
}

// WO[d, n] -> WOt[n, d]
__global__ void pack_WO_kernel(const float* __restrict__ W,
                               unsigned short* __restrict__ Wt, int n) {
    int i = blockIdx.x * blockDim.x + threadIdx.x;
    int stride = gridDim.x * blockDim.x;
    for (; i < n; i += stride) {
        int col = i >> 10;
        int d   = i & 1023;
        Wt[i] = f32_to_bf16(W[(size_t)d * D_MODEL + col]);
    }
}

// -------------------------------------------------------------------- GEMM
// C[MxN] = A[MxK] * Bt[NxK]^T + bias.
// Block = 4 waves -> 64x64 output tile. A/B 64x32 slabs are double-buffered
// in LDS via global_load_async_to_lds_b128; each wave owns 16 rows and
// shares the B slab (4x less global traffic than per-wave fetching).
// storeMode: 0 = bf16 row-major, 1 = bf16 transposed [N,M], 2 = f32 row-major.
__global__ void __launch_bounds__(128)
gemm64x64_bf16(const unsigned short* __restrict__ A,
               const unsigned short* __restrict__ Bt,
               const float* __restrict__ bias,
               unsigned short* __restrict__ Cb,
               float* __restrict__ Cf,
               int M, int N, int K, int storeMode) {
    __shared__ __align__(16) unsigned short Al[2][64 * 32];
    __shared__ __align__(16) unsigned short Bl[2][64 * 32];

    const int wid    = threadIdx.x >> 5;
    const int lane   = threadIdx.x & 31;
    const int t      = threadIdx.x;              // 0..127
    const int ntiles = N >> 6;
    const int row0   = (blockIdx.x / ntiles) << 6;
    const int col0   = (blockIdx.x % ntiles) << 6;

    // Each thread async-copies 2 x 16B of A and 2 x 16B of B per K-slab.
    // chunk ch (0..255): row = ch>>2 (64 rows), k = (ch&3)*8 (32 bf16 per row).
    auto prefetch = [&](int buf, int k0) {
#pragma unroll
        for (int c = 0; c < 2; ++c) {
            const int ch = t + c * 128;
            const int r  = ch >> 2;
            const int kk = (ch & 3) * 8;
            async_load_b128(&Al[buf][r * 32 + kk],
                            A + (size_t)(row0 + r) * K + k0 + kk);
            async_load_b128(&Bl[buf][r * 32 + kk],
                            Bt + (size_t)(col0 + r) * K + k0 + kk);
        }
    };

    prefetch(0, 0);
    v8f acc[4] = {};
    int buf = 0;
    for (int k0 = 0; k0 < K; k0 += 32) {
        const bool more = (k0 + 32) < K;
        if (more) {
            prefetch(buf ^ 1, k0 + 32);
            // in-order ASYNCcnt: <=4 outstanding means current slab landed
            asm volatile("s_wait_asynccnt 0x4" ::: "memory");
        } else {
            asm volatile("s_wait_asynccnt 0x0" ::: "memory");
        }
        __syncthreads();

        v16bf a = load_frag_rowK(&Al[buf][0], 32, wid * 16, 0);
#pragma unroll
        for (int j = 0; j < 4; ++j) {
            v16bf b = load_frag_rowK(&Bl[buf][0], 32, j * 16, 0);
            acc[j] = __builtin_amdgcn_wmma_f32_16x16x32_bf16(
                false, a, false, b, (short)0, acc[j], false, false);
        }
        __syncthreads();
        buf ^= 1;
    }

    const int n = lane & 15, half = lane >> 4;
    const int wrow0 = row0 + wid * 16;
#pragma unroll
    for (int j = 0; j < 4; ++j) {
        const int col = col0 + j * 16 + n;
        const float bv = bias ? bias[col] : 0.0f;
#pragma unroll
        for (int v = 0; v < 8; ++v) {
            const int row = wrow0 + half * 8 + v;
            const float val = acc[j][v] + bv;
            if (storeMode == 0)      Cb[(size_t)row * N + col] = f32_to_bf16(val);
            else if (storeMode == 1) Cb[(size_t)col * M + row] = f32_to_bf16(val);
            else                     Cf[(size_t)row * N + col] = val;
        }
    }
}

// -------------------------------------------------------- flash attention
// q,k : [S, H*64] bf16 row-major ;  vt : [H*64, S] bf16 (K-major over keys)
// heads: [S, H*64] bf16 row-major.  One wave = (head, 16 query rows).
__global__ void __launch_bounds__(128)
attention_kernel(const unsigned short* __restrict__ q,
                 const unsigned short* __restrict__ k,
                 const unsigned short* __restrict__ vt,
                 unsigned short* __restrict__ heads) {
    __shared__ __align__(16) unsigned short pls[4][16 * 32];

    const int wid  = threadIdx.x >> 5;
    const int lane = threadIdx.x & 31;
    const int n = lane & 15, half = lane >> 4;
    const int tid = blockIdx.x * 4 + wid;
    if (tid >= H_HEADS * (S_LEN >> 4)) return;
    const int h  = tid >> 8;          // / (S/16)
    const int r0 = (tid & 255) << 4;  // query row base
    const int hc = h * DKV;

    // query A-fragments (DK=64 -> two K=32 fragments), loaded once
    const v16bf aq0 = load_frag_rowK(q, D_MODEL, r0, hc);
    const v16bf aq1 = load_frag_rowK(q, D_MODEL, r0, hc + 32);

    v8f o[4] = {};
    float mrun[8], lrun[8];
#pragma unroll
    for (int v = 0; v < 8; ++v) { mrun[v] = -3.0e38f; lrun[v] = 0.0f; }

    const int jbmax = (r0 + 15) >> 5;       // causal: keys <= r0+15
    for (int jb = 0; jb <= jbmax; ++jb) {
        const int j0 = jb << 5;

        // scores: 16 rows x 32 keys  (two 16x16 C tiles, 4 WMMAs)
        v8f s[2] = {};
#pragma unroll
        for (int c = 0; c < 2; ++c) {
            v16bf bk0 = load_frag_rowK(k, D_MODEL, j0 + c * 16, hc);
            s[c] = __builtin_amdgcn_wmma_f32_16x16x32_bf16(
                false, aq0, false, bk0, (short)0, s[c], false, false);
            v16bf bk1 = load_frag_rowK(k, D_MODEL, j0 + c * 16, hc + 32);
            s[c] = __builtin_amdgcn_wmma_f32_16x16x32_bf16(
                false, aq1, false, bk1, (short)0, s[c], false, false);
        }

        // online softmax (double causal mask faithfully: 2 * -1e9)
        float p0[8], p1[8];
#pragma unroll
        for (int v = 0; v < 8; ++v) {
            const int rowg = r0 + half * 8 + v;
            float s0 = s[0][v] * 0.125f + (((j0 + n)      > rowg) ? -2.0e9f : 0.0f);
            float s1 = s[1][v] * 0.125f + (((j0 + 16 + n) > rowg) ? -2.0e9f : 0.0f);
            float mv = fmaxf(s0, s1);
            mv = fmaxf(mv, __shfl_xor(mv, 1, 32));
            mv = fmaxf(mv, __shfl_xor(mv, 2, 32));
            mv = fmaxf(mv, __shfl_xor(mv, 4, 32));
            mv = fmaxf(mv, __shfl_xor(mv, 8, 32));
            const float newm = fmaxf(mrun[v], mv);
            const float fsc  = __expf(mrun[v] - newm);
            p0[v] = __expf(s0 - newm);
            p1[v] = __expf(s1 - newm);
            float rs = p0[v] + p1[v];
            rs += __shfl_xor(rs, 1, 32);
            rs += __shfl_xor(rs, 2, 32);
            rs += __shfl_xor(rs, 4, 32);
            rs += __shfl_xor(rs, 8, 32);
            lrun[v] = lrun[v] * fsc + rs;
            mrun[v] = newm;
#pragma unroll
            for (int j = 0; j < 4; ++j) o[j][v] *= fsc;
        }

        // P (C-layout f32) -> LDS bf16 [16 rows x 32 keys] -> A-fragment
#pragma unroll
        for (int v = 0; v < 8; ++v) {
            const int rl = half * 8 + v;
            pls[wid][rl * 32 + n]      = f32_to_bf16(p0[v]);
            pls[wid][rl * 32 + 16 + n] = f32_to_bf16(p1[v]);
        }
        asm volatile("s_wait_dscnt 0" ::: "memory");
        const v16bf pa = load_frag_rowK(&pls[wid][0], 32, 0, 0);

        // O += P @ V   (V B-fragments contiguous along keys via vt)
#pragma unroll
        for (int j = 0; j < 4; ++j) {
            v16bf bv = load_frag_rowK(vt, S_LEN, hc + j * 16, j0);
            o[j] = __builtin_amdgcn_wmma_f32_16x16x32_bf16(
                false, pa, false, bv, (short)0, o[j], false, false);
        }
    }

    // normalize + store concatenated heads [S, H*64] bf16
#pragma unroll
    for (int j = 0; j < 4; ++j)
#pragma unroll
        for (int v = 0; v < 8; ++v) {
            const int rowg = r0 + half * 8 + v;
            heads[(size_t)rowg * D_MODEL + hc + j * 16 + n] =
                f32_to_bf16(o[j][v] / lrun[v]);
        }
}

// ------------------------------------------------------------------ driver
extern "C" void kernel_launch(void* const* d_in, const int* in_sizes, int n_in,
                              void* d_out, int out_size, void* d_ws, size_t ws_size,
                              hipStream_t stream) {
    (void)in_sizes; (void)n_in; (void)out_size; (void)ws_size;

    const float* V  = (const float*)d_in[0];
    const float* K  = (const float*)d_in[1];
    const float* Q  = (const float*)d_in[2];
    const float* WQ = (const float*)d_in[3];
    const float* bQ = (const float*)d_in[4];
    const float* WK = (const float*)d_in[5];
    const float* bK = (const float*)d_in[6];
    const float* WV = (const float*)d_in[7];
    const float* bV = (const float*)d_in[8];
    const float* WO = (const float*)d_in[9];
    const float* bO = (const float*)d_in[10];
    float* out = (float*)d_out;

    const size_t SD = (size_t)S_LEN * D_MODEL;     // 4M elements
    const size_t DD = (size_t)D_MODEL * D_MODEL;   // 1M elements

    char* ws = (char*)d_ws;
    unsigned short* Qb    = (unsigned short*)ws; ws += SD * 2;
    unsigned short* Kb    = (unsigned short*)ws; ws += SD * 2;
    unsigned short* Vb    = (unsigned short*)ws; ws += SD * 2;
    unsigned short* WqT   = (unsigned short*)ws; ws += DD * 2;
    unsigned short* WkT   = (unsigned short*)ws; ws += DD * 2;
    unsigned short* WvT   = (unsigned short*)ws; ws += DD * 2;
    unsigned short* WoT   = (unsigned short*)ws; ws += DD * 2;
    unsigned short* qbuf  = (unsigned short*)ws; ws += SD * 2;
    unsigned short* kbuf  = (unsigned short*)ws; ws += SD * 2;
    unsigned short* vtbuf = (unsigned short*)ws; ws += SD * 2;
    unsigned short* hbuf  = (unsigned short*)ws; ws += SD * 2;

    // 1) dtype conversion + weight repack
    cvt_f32_to_bf16_kernel<<<2048, 256, 0, stream>>>(Q, Qb, (int)SD);
    cvt_f32_to_bf16_kernel<<<2048, 256, 0, stream>>>(K, Kb, (int)SD);
    cvt_f32_to_bf16_kernel<<<2048, 256, 0, stream>>>(V, Vb, (int)SD);
    pack_headW_kernel<<<1024, 256, 0, stream>>>(WQ, WqT, (int)DD);
    pack_headW_kernel<<<1024, 256, 0, stream>>>(WK, WkT, (int)DD);
    pack_headW_kernel<<<1024, 256, 0, stream>>>(WV, WvT, (int)DD);
    pack_WO_kernel<<<1024, 256, 0, stream>>>(WO, WoT, (int)DD);

    // 2) input projections (4096x1024x1024 GEMMs); V stored transposed
    const int gemmBlocks = (S_LEN / 64) * (D_MODEL / 64);  // 1024
    gemm64x64_bf16<<<gemmBlocks, 128, 0, stream>>>(Qb, WqT, bQ, qbuf, nullptr,
                                                   S_LEN, D_MODEL, D_MODEL, 0);
    gemm64x64_bf16<<<gemmBlocks, 128, 0, stream>>>(Kb, WkT, bK, kbuf, nullptr,
                                                   S_LEN, D_MODEL, D_MODEL, 0);
    gemm64x64_bf16<<<gemmBlocks, 128, 0, stream>>>(Vb, WvT, bV, vtbuf, nullptr,
                                                   S_LEN, D_MODEL, D_MODEL, 1);

    // 3) causal flash attention, all heads
    const int attnBlocks = H_HEADS * (S_LEN / 16) / 4;     // 1024
    attention_kernel<<<attnBlocks, 128, 0, stream>>>(qbuf, kbuf, vtbuf, hbuf);

    // 4) output projection -> fp32
    gemm64x64_bf16<<<gemmBlocks, 128, 0, stream>>>(hbuf, WoT, bO, nullptr, out,
                                                   S_LEN, D_MODEL, D_MODEL, 2);
}
